// BiSAGE_53996328845505
// MI455X (gfx1250) — matrix operand order repaired and checked by reading, not compiled, not run on hardware
//
#include <hip/hip_runtime.h>

typedef __attribute__((ext_vector_type(16))) _Float16 v16h;
typedef __attribute__((ext_vector_type(8)))  float    v8f;

#define N_NODES 100000
#define N_EDGESL 3200000LL

// ---------------- degree (float atomic counts; reused for both layers) ----------
__global__ void BiSAGE_deg_kernel(const long long* __restrict__ dst,
                                  float* __restrict__ deg, long long nE) {
    long long e = (long long)blockIdx.x * blockDim.x + threadIdx.x;
    if (e < nE) unsafeAtomicAdd(&deg[dst[e]], 1.0f);
}

__global__ void BiSAGE_inv_kernel(float* __restrict__ deg, int n) {
    int i = blockIdx.x * blockDim.x + threadIdx.x;
    if (i < n) deg[i] = 1.0f / fmaxf(deg[i], 1.0f);
}

// ---------------- convert + pack weights f32 -> f16 WMMA B-fragment order -------
// Packed layout per matrix: frag[(nt*2 + khalf)*32 + lane][i], 16 contiguous halves
// per lane, element is W[(khalf*32 + (i>>3)*16 + (lane>>4)*8 + (i&7))*LDB + nt*16 + (lane&15)].
// So each B fragment in the combine kernel is one aligned 32-byte v16h load.
__global__ void BiSAGE_packw_kernel(const float* __restrict__ W1l, const float* __restrict__ W1r,
                                    const float* __restrict__ W2l, const float* __restrict__ W2r,
                                    _Float16* __restrict__ out) {
    int t = blockIdx.x * blockDim.x + threadIdx.x;      // 0..12287
    if (t >= 12288) return;
    const float* W; int ntiles; _Float16* o; int base;
    if (t < 4096)       { W = W1l; ntiles = 4; o = out;         base = t; }
    else if (t < 8192)  { W = W1r; ntiles = 4; o = out + 4096;  base = t - 4096; }
    else if (t < 10240) { W = W2l; ntiles = 2; o = out + 8192;  base = t - 8192; }
    else                { W = W2r; ntiles = 2; o = out + 10240; base = t - 10240; }
    int ldb  = ntiles * 16;
    int i    = base & 15;
    int lane = (base >> 4) & 31;
    int kh   = (base >> 9) & 1;
    int nt   = base >> 10;
    int hi   = lane >> 4;
    int n    = nt * 16 + (lane & 15);
    int k    = kh * 32 + (i >> 3) * 16 + hi * 8 + (i & 7);
    o[base]  = (_Float16)W[k * ldb + n];
}

// ---------------- edge scatter: sum[dst] += feat[src], float4 granularity -------
__global__ void BiSAGE_scatter_kernel(const float* __restrict__ feat,
                                      const long long* __restrict__ src,
                                      const long long* __restrict__ dst,
                                      float* __restrict__ sum, long long nE) {
    long long gid = (long long)blockIdx.x * blockDim.x + threadIdx.x;
    long long e = gid >> 4;
    if (e >= nE) return;
    int q = (int)(gid & 15);
    long long s = src[e], d = dst[e];
    const float4 v = *(const float4*)(feat + s * 64 + q * 4);
    float* o = sum + d * 64 + q * 4;
    unsafeAtomicAdd(o + 0, v.x);
    unsafeAtomicAdd(o + 1, v.y);
    unsafeAtomicAdd(o + 2, v.z);
    unsafeAtomicAdd(o + 3, v.w);
}

// ---------------- fused combine: out = [relu]( (sum*inv)@Wl + b + feat@Wr ) -----
// One wave per 16-row node block; K = 64 (two 16x32 f16 A tiles per operand).
// Wl/Wr point at pre-packed B fragments (see BiSAGE_packw_kernel).
template <int NTILES, bool RELU>
__global__ __launch_bounds__(256) void BiSAGE_combine_kernel(
    const float* __restrict__ feat, const float* __restrict__ sum,
    const float* __restrict__ inv,
    const _Float16* __restrict__ Wl, const _Float16* __restrict__ Wr,
    const float* __restrict__ bias, float* __restrict__ out) {
    const int LDB = NTILES * 16;
    int lane = threadIdx.x & 31;
    int wave = threadIdx.x >> 5;
    int rowblock = blockIdx.x * 8 + wave;
    if (rowblock >= N_NODES / 16) return;   // wave-uniform: EXEC stays all-1s for WMMA
    int m0 = rowblock * 16;
    int lid = lane & 15;                    // A row / B,C,D column within tile
    int hi  = lane >> 4;                    // lane-half selector

    const float* frow = feat + (long long)(m0 + lid) * 64;
    const float* srow = sum  + (long long)(m0 + lid) * 64;
    float scale = inv[m0 + lid];

    // A-matrix 16x32 f16 layout: lanes 0-15 hold K=[0..7]+[16..23], lanes 16-31 K=[8..15]+[24..31]
    v16h ax_lo, ax_hi, ag_lo, ag_hi;
    int kb = hi * 8;
#pragma unroll
    for (int i = 0; i < 8; i++) {
        ax_lo[i]     = (_Float16)frow[kb + i];
        ax_lo[i + 8] = (_Float16)frow[kb + 16 + i];
        ax_hi[i]     = (_Float16)frow[32 + kb + i];
        ax_hi[i + 8] = (_Float16)frow[32 + kb + 16 + i];
        ag_lo[i]     = (_Float16)(srow[kb + i] * scale);
        ag_lo[i + 8] = (_Float16)(srow[kb + 16 + i] * scale);
        ag_hi[i]     = (_Float16)(srow[32 + kb + i] * scale);
        ag_hi[i + 8] = (_Float16)(srow[32 + kb + 16 + i] * scale);
    }

#pragma unroll
    for (int nt = 0; nt < NTILES; nt++) {
        // pre-packed B fragments: one aligned 32B v16h load each
        const _Float16* WlF = Wl + ((nt * 2) * 32 + lane) * 16;
        const _Float16* WrF = Wr + ((nt * 2) * 32 + lane) * 16;
        v16h bl_lo = *(const v16h*)(WlF);
        v16h bl_hi = *(const v16h*)(WlF + 512);   // khalf=1 fragment block
        v16h br_lo = *(const v16h*)(WrF);
        v16h br_hi = *(const v16h*)(WrF + 512);

        float bv = bias[nt * 16 + lid];
        v8f c;
#pragma unroll
        for (int j = 0; j < 8; j++) c[j] = bv;
        c = __builtin_amdgcn_wmma_f32_16x16x32_f16(false, ag_lo, false, bl_lo, (short)0, c, false, false);
        c = __builtin_amdgcn_wmma_f32_16x16x32_f16(false, ag_hi, false, bl_hi, (short)0, c, false, false);
        c = __builtin_amdgcn_wmma_f32_16x16x32_f16(false, ax_lo, false, br_lo, (short)0, c, false, false);
        c = __builtin_amdgcn_wmma_f32_16x16x32_f16(false, ax_hi, false, br_hi, (short)0, c, false, false);
#pragma unroll
        for (int j = 0; j < 8; j++) {
            float v = RELU ? fmaxf(c[j], 0.0f) : c[j];
            out[(long long)(m0 + j + hi * 8) * LDB + nt * 16 + lid] = v;  // D: M = j + 8*hi
        }
    }
}

extern "C" void kernel_launch(void* const* d_in, const int* in_sizes, int n_in,
                              void* d_out, int out_size, void* d_ws, size_t ws_size,
                              hipStream_t stream) {
    const float*     x   = (const float*)d_in[0];
    const long long* ei  = (const long long*)d_in[1];   // int64 [2, E]
    const long long* src = ei;
    const long long* dst = ei + N_EDGESL;
    const float* W1l = (const float*)d_in[2];
    const float* b1  = (const float*)d_in[3];
    const float* W1r = (const float*)d_in[4];
    const float* W2l = (const float*)d_in[5];
    const float* b2  = (const float*)d_in[6];
    const float* W2r = (const float*)d_in[7];
    float* out = (float*)d_out;

    char* ws = (char*)d_ws;
    size_t off = 0;
    float* sum = (float*)(ws + off); off += (size_t)N_NODES * 64 * sizeof(float);          // 25.6 MB
    float* inv = (float*)(ws + off); off += (((size_t)N_NODES * sizeof(float)) + 255) & ~(size_t)255;
    float* h   = (float*)(ws + off); off += (size_t)N_NODES * 64 * sizeof(float);          // 25.6 MB
    _Float16* wf  = (_Float16*)(ws + off);                                                 // 24 KB packed
    _Float16* w1l = wf, * w1r = wf + 4096, * w2l = wf + 8192, * w2r = wf + 10240;

    const int TPB = 256;
    const int degBlocks     = (int)((N_EDGESL + TPB - 1) / TPB);
    const int scatterBlocks = (int)((N_EDGESL * 16 + TPB - 1) / TPB);
    const int combineBlocks = (N_NODES / 16 + 7) / 8;

    // layer 1
    hipMemsetAsync(sum, 0, (size_t)N_NODES * 64 * sizeof(float), stream);
    hipMemsetAsync(inv, 0, (size_t)N_NODES * sizeof(float), stream);
    BiSAGE_packw_kernel<<<48, TPB, 0, stream>>>(W1l, W1r, W2l, W2r, wf);
    BiSAGE_deg_kernel<<<degBlocks, TPB, 0, stream>>>(dst, inv, N_EDGESL);
    BiSAGE_inv_kernel<<<(N_NODES + TPB - 1) / TPB, TPB, 0, stream>>>(inv, N_NODES);
    BiSAGE_scatter_kernel<<<scatterBlocks, TPB, 0, stream>>>(x, src, dst, sum, N_EDGESL);
    BiSAGE_combine_kernel<4, true><<<combineBlocks, TPB, 0, stream>>>(x, sum, inv, w1l, w1r, b1, h);

    // layer 2 (same degrees / inv)
    hipMemsetAsync(sum, 0, (size_t)N_NODES * 64 * sizeof(float), stream);
    BiSAGE_scatter_kernel<<<scatterBlocks, TPB, 0, stream>>>(h, src, dst, sum, N_EDGESL);
    BiSAGE_combine_kernel<2, false><<<combineBlocks, TPB, 0, stream>>>(h, sum, inv, w2l, w2r, b2, out);
}
